// GraphNeuralNetwork_10754598109368
// MI455X (gfx1250) — matrix operand order repaired and checked by reading, not compiled, not run on hardware
//
#include <hip/hip_runtime.h>
#include <hip/hip_bf16.h>
#include <math.h>

typedef __attribute__((ext_vector_type(16))) _Float16 v16h;
typedef __attribute__((ext_vector_type(8)))  _Float16 v8h;
typedef __attribute__((ext_vector_type(8)))  float    v8f;

#define NODES   1024
#define DIM     128
#define NRBF    16
#define NLAYERS 3
#define CUTOFF  2.0f
#define GAMMA_  10.0f
#define PI_F    3.14159265358979323846f

// ---------------------------------------------------------------------------
// Kernel 0: h = embed (f32, running state in d_out), hT[d][i] = f16(embed[i][d])
// ---------------------------------------------------------------------------
__global__ void k_init(const float* __restrict__ embed,
                       float* __restrict__ h,
                       _Float16* __restrict__ hT) {
    int i = blockIdx.x;          // node
    int d = threadIdx.x;         // feature
    float v = embed[i * DIM + d];
    h[i * DIM + d] = v;
    hT[d * NODES + i] = (_Float16)v;
}

// ---------------------------------------------------------------------------
// Kernel 1: rbf[k][i][j] (f16).  One thread per (i,j) pair.
// ---------------------------------------------------------------------------
__global__ void k_rbf(const float* __restrict__ r,
                      _Float16* __restrict__ rbf) {
    int idx = blockIdx.x * blockDim.x + threadIdx.x;   // 0 .. 1024*1024-1
    int i = idx >> 10;
    int j = idx & 1023;

    float dx = r[i * 3 + 0] - r[j * 3 + 0];
    float dy = r[i * 3 + 1] - r[j * 3 + 1];
    float dz = r[i * 3 + 2] - r[j * 3 + 2];
    float d2 = dx * dx + dy * dy + dz * dz;
    bool  diag = (i == j);
    float dist = sqrtf(diag ? 1.0f : d2);
    bool  mask = (dist < CUTOFF) && !diag;
    float fcut = 0.5f * (__cosf(PI_F * dist / CUTOFF) + 1.0f);
    float w = mask ? fcut : 0.0f;

    const float dmu = CUTOFF / (float)(NRBF - 1);
#pragma unroll
    for (int k = 0; k < NRBF; ++k) {
        float t = dist - (float)k * dmu;
        float v = __expf(-GAMMA_ * t * t) * w;
        rbf[(size_t)k * NODES * NODES + (size_t)idx] = (_Float16)v;
    }
}

// ---------------------------------------------------------------------------
// Async fill of one LDS B-buffer: smem[buf] <= hT[0..127][kk..kk+32)
// Chunk = 128 rows x 64 B = 8 KB; 256 threads x 2 x b128 ops each.
// Tracked on ASYNCcnt (gfx1250 async-to-LDS path).
// ---------------------------------------------------------------------------
__device__ __forceinline__ void fill_async(const _Float16* hT, unsigned ldsBase,
                                           int kk, int tid) {
#pragma unroll
    for (int r = 0; r < 2; ++r) {
        int o16 = r * 256 + tid;             // 16-byte unit, 0..511
        int d   = o16 >> 2;                  // row (feature) index
        int ju  = (o16 & 3) * 8;             // half offset within row
        unsigned lds  = ldsBase + (unsigned)o16 * 16u;
        unsigned voff = (unsigned)((d * NODES + kk + ju) * 2);
        asm volatile("global_load_async_to_lds_b128 %0, %1, %2"
                     :: "v"(lds), "v"(voff),
                        "s"((unsigned long long)(size_t)hT)
                     : "memory");
    }
}

// ---------------------------------------------------------------------------
// Kernel 2: batched GEMM via WMMA, LDS-staged B with async double buffering.
//   G[k][i][d] = sum_j rbf[k][i][j] * h[j][d]
// grid = (M-tiles = 8, k = 16); block = 256 (8 waves).
// ---------------------------------------------------------------------------
__global__ void __launch_bounds__(256)
k_gemm(const _Float16* __restrict__ rbf,
       const _Float16* __restrict__ hT,    // [DIM][NODES], f16
       float* __restrict__ G) {
    const int k     = blockIdx.y;
    const int Mbase = blockIdx.x * 128;
    const int tid   = threadIdx.x;
    const int wave  = tid >> 5;
    const int lane  = tid & 31;
    const int lo16  = lane & 15;
    const int hi    = lane >> 4;             // 0 or 1

    __shared__ _Float16 smemB[2][DIM][32];   // 2 x 8 KB double buffer

    const _Float16* A    = rbf + (size_t)k * NODES * NODES;
    const _Float16* Arow = A + (size_t)(Mbase + wave * 16 + lo16) * NODES;
    const int akb = hi * 8;                  // A K sub-offset (halves)

    const unsigned ldsB[2] = { (unsigned)(size_t)&smemB[0][0][0],
                               (unsigned)(size_t)&smemB[1][0][0] };

    v8f acc[8];
#pragma unroll
    for (int nt = 0; nt < 8; ++nt) acc[nt] = (v8f){0,0,0,0,0,0,0,0};

    union AU { v16h v; v8h h2[2]; };
    union BU { v16h v; v8h h2[2]; };

    // prologue: async-fill buffer 0, preload first A fragment
    fill_async(hT, ldsB[0], 0, tid);
    AU ua[2];
    ua[0].h2[0] = *(const v8h*)(Arow + 0 + akb);
    ua[0].h2[1] = *(const v8h*)(Arow + 16 + akb);

    int p = 0;
    for (int kk = 0; kk < NODES; kk += 32, p ^= 1) {
        const bool more = (kk + 32) < NODES;
        if (more) {
            // software pipeline: next B chunk (async, ASYNCcnt) + next A (regs)
            fill_async(hT, ldsB[p ^ 1], kk + 32, tid);
            ua[p ^ 1].h2[0] = *(const v8h*)(Arow + kk + 32 + akb);
            ua[p ^ 1].h2[1] = *(const v8h*)(Arow + kk + 48 + akb);
            __builtin_prefetch(Arow + kk + 64, 0, 3);
            // 2 newest async ops may stay outstanding; older (current buf) done
            asm volatile("s_wait_asynccnt 0x2" ::: "memory");
        } else {
            asm volatile("s_wait_asynccnt 0x0" ::: "memory");
        }
        __syncthreads();   // all waves' fills of buffer p visible

        // read 8 B fragments from LDS (ds_load_b128 x2 each)
        BU ub[8];
#pragma unroll
        for (int nt = 0; nt < 8; ++nt) {
            const _Float16* Bp = &smemB[p][nt * 16 + lo16][hi * 8];
            ub[nt].h2[0] = *(const v8h*)(Bp);
            ub[nt].h2[1] = *(const v8h*)(Bp + 16);
        }
        __syncthreads();   // reads done before buffer p is refilled later

#pragma unroll
        for (int nt = 0; nt < 8; ++nt)
            acc[nt] = __builtin_amdgcn_wmma_f32_16x16x32_f16(
                false, ua[p].v, false, ub[nt].v, (short)0, acc[nt], false, false);
    }

    // --- store C: lane (hi,lo16), element e -> row M = e + 8*hi, col N = lo16 ---
    const int mrow = hi * 8;
#pragma unroll
    for (int nt = 0; nt < 8; ++nt) {
        const int dcol = nt * 16 + lo16;
#pragma unroll
        for (int e = 0; e < 8; ++e) {
            const int irow = Mbase + wave * 16 + mrow + e;
            G[((size_t)k * NODES + irow) * DIM + dcol] = acc[nt][e];
        }
    }
}

// ---------------------------------------------------------------------------
// Kernel 3: per-node combine + residual MLP (f32), refresh hT (f16).
// ---------------------------------------------------------------------------
__global__ void __launch_bounds__(128)
k_combine(const float* __restrict__ G,
          const float* __restrict__ w_filt_l,   // [NRBF][DIM]
          const float* __restrict__ w1_l,       // [DIM][DIM]
          const float* __restrict__ b1_l,       // [DIM]
          const float* __restrict__ w2_l,       // [DIM][DIM]
          const float* __restrict__ b2_l,       // [DIM]
          float* __restrict__ h,
          _Float16* __restrict__ hT) {
    const int i = blockIdx.x;
    const int d = threadIdx.x;

    __shared__ float s_agg[DIM];
    __shared__ float s_y1[DIM];

    float agg = 0.0f;
#pragma unroll
    for (int k = 0; k < NRBF; ++k)
        agg += w_filt_l[k * DIM + d] * G[((size_t)k * NODES + i) * DIM + d];
    s_agg[d] = agg;
    __syncthreads();

    float y = b1_l[d];
#pragma unroll 8
    for (int e = 0; e < DIM; ++e)
        y += s_agg[e] * w1_l[e * DIM + d];
    float sil = y / (1.0f + __expf(-y));      // silu
    s_y1[d] = sil;
    __syncthreads();

    float m = b2_l[d];
#pragma unroll 8
    for (int e = 0; e < DIM; ++e)
        m += s_y1[e] * w2_l[e * DIM + d];

    float hn = h[i * DIM + d] + m;
    h[i * DIM + d] = hn;
    hT[d * NODES + i] = (_Float16)hn;
}

// ---------------------------------------------------------------------------
extern "C" void kernel_launch(void* const* d_in, const int* in_sizes, int n_in,
                              void* d_out, int out_size, void* d_ws, size_t ws_size,
                              hipStream_t stream) {
    const float* r      = (const float*)d_in[0];   // [1024,3]
    const float* embed  = (const float*)d_in[1];   // [1024,128]
    const float* w_filt = (const float*)d_in[2];   // [3,16,128]
    const float* w1     = (const float*)d_in[3];   // [3,128,128]
    const float* b1     = (const float*)d_in[4];   // [3,128]
    const float* w2     = (const float*)d_in[5];   // [3,128,128]
    const float* b2     = (const float*)d_in[6];   // [3,128]

    float* h = (float*)d_out;                      // running node state [1024,128]

    // workspace carve-up
    char* ws = (char*)d_ws;
    _Float16* rbf = (_Float16*)(ws);                                    // 32 MB
    _Float16* hT  = (_Float16*)(ws + (size_t)NRBF * NODES * NODES * 2); // 256 KB
    float*    G   = (float*)(ws + (size_t)NRBF * NODES * NODES * 2
                                + (size_t)DIM * NODES * 2);             // 8 MB

    k_init<<<NODES, DIM, 0, stream>>>(embed, h, hT);
    k_rbf <<<(NODES * NODES) / 256, 256, 0, stream>>>(r, rbf);

    for (int l = 0; l < NLAYERS; ++l) {
        k_gemm<<<dim3(NODES / 128, NRBF), 256, 0, stream>>>(rbf, hT, G);
        k_combine<<<NODES, DIM, 0, stream>>>(
            G,
            w_filt + (size_t)l * NRBF * DIM,
            w1     + (size_t)l * DIM * DIM,
            b1     + (size_t)l * DIM,
            w2     + (size_t)l * DIM * DIM,
            b2     + (size_t)l * DIM,
            h, hT);
    }
}
